// CRFLayer_82489141887691
// MI455X (gfx1250) — matrix operand order repaired and checked by reading, not compiled, not run on hardware
//
#include <hip/hip_runtime.h>

typedef _Float16 v16h __attribute__((ext_vector_type(16)));
typedef _Float16 v8h  __attribute__((ext_vector_type(8)));
typedef float    v8f  __attribute__((ext_vector_type(8)));

#define T_LEN  512
#define K_TAGS 48
#define KP     50      // transitions row stride (K+2)
#define START_ 48
#define STOP_  49
#define ROWS   16      // batch rows per block (WMMA M)

#define LN2F    0.6931471805599453f
#define INVLN2F 1.4426950408889634f

static __device__ __forceinline__ v16h cat8(v8h lo, v8h hi) {
  return __builtin_shufflevector(lo, hi, 0,1,2,3,4,5,6,7,8,9,10,11,12,13,14,15);
}

// non-returning LDS float max (DS_MAX_NUM_F32)
static __device__ __forceinline__ void ldsMaxF(float* p, float v) {
  __hip_atomic_fetch_max(p, v, __ATOMIC_RELAXED, __HIP_MEMORY_SCOPE_WORKGROUP);
}

__global__ __launch_bounds__(128) void crf_forward_kernel(
    const float* __restrict__ emissions,       // [B,T,48] f32
    const long long* __restrict__ tags,        // [B,T] i64
    const unsigned char* __restrict__ mask,    // [B,T] bool
    const float* __restrict__ trans,           // [50,50] f32
    float* __restrict__ partials)              // [gridDim.x]
{
  __shared__ alignas(16) _Float16 pP[2][ROWS][64]; // ping-pong A: p = 2^(alpha2 - n)
  __shared__ alignas(16) _Float16 Et[64][64];      // Et[n][k] = exp(trans[k][n]), zero-padded
  __shared__ alignas(16) float keyF[2][ROWS];      // running row-max normalizers (parity-striped)
  __shared__ unsigned maskBitsS[2];
  __shared__ float goldS[ROWS];
  __shared__ float red[ROWS];

  const int tid = threadIdx.x;
  const int wv  = tid >> 5;
  const int ln  = tid & 31;
  const int b0  = blockIdx.x * ROWS;

  // ---- setup: E^T tile, both A buffers (one-hot START in buf0), normalizers = 0 ----
  for (int idx = tid; idx < 64 * 64; idx += 128) {
    int n = idx >> 6, k = idx & 63;
    float e = 0.0f;
    if (n < KP && k < KP) e = __expf(trans[k * KP + n]);
    Et[n][k] = (_Float16)e;
  }
  for (int idx = tid; idx < 2 * ROWS * 64; idx += 128) {
    int q = idx >> 10, r = (idx >> 6) & 15, j = idx & 63;
    pP[q][r][j] = (_Float16)((q == 0 && j == START_) ? 1.0f : 0.0f);
  }
  if (tid < ROWS) { keyF[0][tid] = 0.0f; keyF[1][tid] = 0.0f; }

  unsigned char mvCur = 0, mvNext = 0;
  if (wv == 3 && ln < ROWS) {                      // prime mask pipeline (step 0)
    mvCur = mask[(size_t)(b0 + ln) * T_LEN];
    unsigned mb = (unsigned)__ballot(mvCur != 0);
    if (ln == 0) maskBitsS[0] = mb;
  }
  __syncthreads();

  // ---- constant B fragments; waves 0..2 own j-tiles 0..47 ----
  v16h bf0 = {}, bf1 = {};
  const int jcol  = wv * 16 + (ln & 15);
  const int rbase = (ln & 16) ? 8 : 0;
  const int M     = ln & 15;
  const int kh    = (ln & 16) ? 8 : 0;
  if (wv < 3) {
    int kk = (ln & 16) ? 8 : 0;
    bf0 = cat8(*(const v8h*)&Et[jcol][kk],      *(const v8h*)&Et[jcol][kk + 16]);
    bf1 = cat8(*(const v8h*)&Et[jcol][32 + kk], *(const v8h*)&Et[jcol][32 + kk + 16]);
  }

  float ecur[8];
  if (wv < 3) {
#pragma unroll
    for (int v = 0; v < 8; ++v)
      ecur[v] = emissions[((size_t)(b0 + rbase + v) * T_LEN) * K_TAGS + jcol];
  }
  float n_prev[8];
#pragma unroll
  for (int v = 0; v < 8; ++v) n_prev[v] = 0.0f;

  float gold = 0.0f;
  int prevTag = 0, lastTag = 0;

  auto stepBody = [&](int t, int PA, int PB) {
    const unsigned mbits = maskBitsS[PA];
    if (wv < 3) {
      // A fragments from read buffer + GEMM (2 K-steps)
      v16h a0 = cat8(*(const v8h*)&pP[PA][M][kh],      *(const v8h*)&pP[PA][M][kh + 16]);
      v16h a1 = cat8(*(const v8h*)&pP[PA][M][32 + kh], *(const v8h*)&pP[PA][M][32 + kh + 16]);
      v8f acc = {};
      acc = __builtin_amdgcn_wmma_f32_16x16x32_f16(false, a0, false, bf0, (short)0, acc, false, false);
      acc = __builtin_amdgcn_wmma_f32_16x16x32_f16(false, a1, false, bf1, (short)0, acc, false, false);

      // fresh (lag<=2) row normalizers: plain vector loads
      float4 n0 = *(const float4*)&keyF[PA][rbase];
      float4 n1 = *(const float4*)&keyF[PA][rbase + 4];
      float ncur[8] = { n0.x, n0.y, n0.z, n0.w, n1.x, n1.y, n1.z, n1.w };

      // next-step emission loads (a full step of latency to hide)
      float enext[8];
      if (t + 1 < T_LEN) {
#pragma unroll
        for (int v = 0; v < 8; ++v)
          enext[v] = emissions[((size_t)(b0 + rbase + v) * T_LEN + (t + 1)) * K_TAGS + jcol];
      }

      // sc2 = n_prev + log2(sum p*E) + emit/ln2  (med3 clamp, raw v_log_f32)
      float sc[8];
#pragma unroll
      for (int v = 0; v < 8; ++v)
        sc[v] = n_prev[v]
              + __builtin_amdgcn_logf(__builtin_amdgcn_fmed3f(acc[v], 1e-30f, 3.0e38f))
              + ecur[v] * INVLN2F;

      // scalar-uniform fast path when all 16 rows update (the common case)
      unsigned mb16 = (unsigned)__builtin_amdgcn_readfirstlane((int)(mbits & 0xFFFFu));
      if (mb16 == 0xFFFFu) {
#pragma unroll
        for (int v = 0; v < 8; ++v) {
          int r = rbase + v;
          float p = __builtin_amdgcn_fmed3f(
                      __builtin_amdgcn_exp2f(sc[v] - ncur[v]), 0.0f, 60000.0f);
          pP[PB][r][jcol] = (_Float16)p;
          ldsMaxF(&keyF[PB][r], sc[v]);
        }
      } else {
#pragma unroll
        for (int v = 0; v < 8; ++v) {
          int r = rbase + v;
          if ((mbits >> r) & 1u) {
            float p = __builtin_amdgcn_fmed3f(
                        __builtin_amdgcn_exp2f(sc[v] - ncur[v]), 0.0f, 60000.0f);
            pP[PB][r][jcol] = (_Float16)p;
            ldsMaxF(&keyF[PB][r], sc[v]);
          }
        }
      }
#pragma unroll
      for (int v = 0; v < 8; ++v) { n_prev[v] = ncur[v]; ecur[v] = enext[v]; }
    } else if (ln < ROWS) {
      unsigned char mv = mvCur;
      if (t + 1 < T_LEN) {                    // stage next step's mask ballot
        mvNext = mask[(size_t)(b0 + ln) * T_LEN + (t + 1)];
        unsigned mbn = (unsigned)__ballot(mvNext != 0);
        if (ln == 0) maskBitsS[PB] = mbn;
      }
      if (t + 2 < T_LEN) {
        const float* nxt = &emissions[((size_t)(b0 + ln) * T_LEN + (t + 2)) * K_TAGS];
        __builtin_prefetch(nxt, 0, 3);
        __builtin_prefetch(nxt + 32, 0, 3);
      }
      if (mv) {
        pP[PB][ln][START_] = (_Float16)0.0f;  // retire one-hot on update
      } else {
        // frozen row: carry the whole row (cols 0..63) to the write buffer
#pragma unroll
        for (int q = 0; q < 8; ++q)
          *(uint4*)&pP[PB][ln][q * 8] = *(const uint4*)&pP[PA][ln][q * 8];
      }
      // gold-path score
      int tg = (int)tags[(size_t)(b0 + ln) * T_LEN + t];
      if (t == 0) gold += trans[START_ * KP + tg];
      if (mv) {
        gold += emissions[((size_t)(b0 + ln) * T_LEN + t) * K_TAGS + tg];
        if (t > 0) gold += trans[prevTag * KP + tg];
        lastTag = tg;
      }
      prevTag = tg;
      mvCur = mvNext;
    }
    __syncthreads();   // single barrier: pP[PB], keyF[PB], maskBitsS[PB] published
  };

  for (int tt = 0; tt < T_LEN; tt += 2) {
    stepBody(tt,     0, 1);
    stepBody(tt + 1, 1, 0);
  }

  // ---- finish gold, log_z = ln2*(n + log2(sum p*exp(trans[:,STOP]))), block partial ----
  if (wv == 3 && ln < ROWS) {
    gold += trans[lastTag * KP + STOP_];
    goldS[ln] = gold;
  }
  __syncthreads();
  if (tid < ROWS) {
    float s = 0.0f;
#pragma unroll 4
    for (int j = 0; j < K_TAGS; ++j)
      s += (float)pP[0][tid][j] * __expf(trans[j * KP + STOP_]);   // final buffer = buf0 (T even)
    // normalizer used at t=511 was keyF[1] (read parity), untouched afterwards
    float logz = LN2F * (keyF[1][tid]
               + __builtin_amdgcn_logf(__builtin_amdgcn_fmed3f(s, 1e-30f, 3.0e38f)));
    red[tid] = logz - goldS[tid];
  }
  __syncthreads();
  if (tid == 0) {
    float s = 0.0f;
    for (int r = 0; r < ROWS; ++r) s += red[r];
    partials[blockIdx.x] = s;
  }
}

__global__ void crf_finalize_kernel(const float* __restrict__ partials, int nblocks,
                                    float invB, float* __restrict__ out)
{
  float s = 0.0f;
  for (int i = 0; i < nblocks; ++i) s += partials[i];
  float nll = s * invB;
  out[0] = 0.9f * nll + 0.1f * __logf((float)K_TAGS);
}

extern "C" void kernel_launch(void* const* d_in, const int* in_sizes, int n_in,
                              void* d_out, int out_size, void* d_ws, size_t ws_size,
                              hipStream_t stream) {
  const float*         emissions = (const float*)d_in[0];
  const long long*     tags      = (const long long*)d_in[1];
  const unsigned char* mask      = (const unsigned char*)d_in[2];
  const float*         trans     = (const float*)d_in[3];
  float* out      = (float*)d_out;
  float* partials = (float*)d_ws;

  int B = in_sizes[1] / T_LEN;      // tags is [B, T]
  int nblocks = B / ROWS;           // 1024/16 = 64

  crf_forward_kernel<<<nblocks, 128, 0, stream>>>(emissions, tags, mask, trans, partials);
  crf_finalize_kernel<<<1, 1, 0, stream>>>(partials, nblocks, 1.0f / (float)B, out);
}
